// xLSTMForecast_33672543600774
// MI455X (gfx1250) — matrix operand order repaired and checked by reading, not compiled, not run on hardware
//
#include <hip/hip_runtime.h>
#include <hip/hip_bf16.h>

// ---------------------------------------------------------------------------
// Types for CDNA5 WMMA
// ---------------------------------------------------------------------------
typedef __bf16 v16bf __attribute__((ext_vector_type(16)));
typedef float  v8f   __attribute__((ext_vector_type(8)));
typedef unsigned int u32x4 __attribute__((ext_vector_type(4)));

union Frag32B {            // 32 bytes: one WMMA A/B fragment per lane
    u32x4 q[2];
    v16bf v;
};

// ---------------------------------------------------------------------------
// Problem dims
// ---------------------------------------------------------------------------
#define BB   512          // batch
#define SS   128          // sequence
#define DD   64           // input dim
#define HH   512          // hidden
#define N4H  2048         // 4*H
#define EPSL 1e-5f

// ---------------------------------------------------------------------------
// gfx1250 async global->LDS helpers (ASYNCcnt-tracked, bypass VGPRs)
// ---------------------------------------------------------------------------
__device__ __forceinline__ unsigned lds_addr(const void* p)
{
    // generic -> LDS(AS3) -> 32-bit LDS byte address
    return (unsigned)(unsigned long long)
        (__attribute__((address_space(3))) const void*)p;
}

__device__ __forceinline__ void async_ld16(unsigned lds, const void* g)
{
    unsigned long long ga = (unsigned long long)(size_t)g;
    asm volatile("global_load_async_to_lds_b128 %0, %1, off"
                 :: "v"(lds), "v"(ga) : "memory");
}

__device__ __forceinline__ void wait_async0()
{
    asm volatile("s_wait_asynccnt 0x0" ::: "memory");
}

// ---------------------------------------------------------------------------
// Prep kernels (run once per launch; deterministic)
// ---------------------------------------------------------------------------
__global__ void cvt_w_kernel(const float* __restrict__ W, __bf16* __restrict__ Wt,
                             int K, int N, int total)
{
    int idx = blockIdx.x * 256 + threadIdx.x;
    if (idx >= total) return;
    int n = idx / K;
    int k = idx - n * K;
    Wt[idx] = (__bf16)W[(size_t)k * N + n];   // Wt[n][k] = W[k][n]
}

__global__ void cvt_x_kernel(const float* __restrict__ x, __bf16* __restrict__ xt)
{
    int idx = blockIdx.x * 256 + threadIdx.x;        // over S*B*D
    if (idx >= SS * BB * DD) return;
    int t   = idx / (BB * DD);
    int rem = idx - t * (BB * DD);
    int b   = rem >> 6;           // /DD
    int d   = rem & 63;           // %DD
    xt[idx] = (__bf16)x[((size_t)b * SS + t) * DD + d];   // xt[t][b][d]
}

__global__ void zero_kernel(unsigned int* __restrict__ p, int n_dwords)
{
    int i = blockIdx.x * 256 + threadIdx.x;
    if (i < n_dwords) p[i] = 0u;
}

// ---------------------------------------------------------------------------
// bf16 GEMM:  Z[M=512][N=2048] = concat(A1[M][K1], A2[M][K2]) * Wt^T + bias
// Wt stored as [N][Ktot] bf16 (pre-transposed) so per-lane B loads are
// contiguous along K.  Workgroup tile 64x128, K-chunk 32.
// Async-to-LDS double-buffered pipeline: one barrier + one asynccnt wait per
// chunk; loads for chunk i+1 overlap WMMA of chunk i.
// 8 waves, each wave owns a 32x32 tile -> 4x v_wmma_f32_16x16x32_bf16/chunk.
// ---------------------------------------------------------------------------
__global__ void __launch_bounds__(256)
gemm_bf16_kernel(const __bf16* __restrict__ A1, int lda1, int K1,
                 const __bf16* __restrict__ A2, int lda2,
                 const __bf16* __restrict__ Wt,
                 const float*  __restrict__ bias,
                 float* __restrict__ Z, int Ktot)
{
    __shared__ __bf16 Abuf[2][64 * 32];      // [row][k]   2 x 4 KB
    __shared__ __bf16 Bbuf[2][128 * 32];     // [col][k]   2 x 8 KB

    const int tid  = threadIdx.x;
    const int lane = tid & 31;
    const int wid  = tid >> 5;
    const int ll   = lane & 15;
    const int lh   = lane >> 4;
    const int wave_m = wid & 1;              // 2 x 32 rows
    const int wave_n = wid >> 1;             // 4 x 32 cols
    const int c4 = tid & 3;                  // which 16B of a 64B k-row
    const int rA = tid >> 2;                 // 0..63
    const int mBase = blockIdx.y * 64;
    const int nBase = blockIdx.x * 128;

    // per-thread source bases
    const __bf16* a1p = A1 + (size_t)(mBase + rA) * lda1 + c4 * 8;
    const __bf16* a2p = A2 + (size_t)(mBase + rA) * lda2 + c4 * 8;
    const __bf16* w0p = Wt + (size_t)(nBase + rA) * Ktot + c4 * 8;
    const __bf16* w1p = Wt + (size_t)(nBase + rA + 64) * Ktot + c4 * 8;

    // per-thread LDS destinations (both buffers)
    unsigned ldsA[2], ldsB0[2], ldsB1[2];
    #pragma unroll
    for (int b = 0; b < 2; ++b) {
        ldsA[b]  = lds_addr(&Abuf[b][rA * 32 + c4 * 8]);
        ldsB0[b] = lds_addr(&Bbuf[b][rA * 32 + c4 * 8]);
        ldsB1[b] = lds_addr(&Bbuf[b][(rA + 64) * 32 + c4 * 8]);
    }

    v8f acc[2][2];
    acc[0][0] = {}; acc[0][1] = {}; acc[1][0] = {}; acc[1][1] = {};

    // issue chunk kk into buffer buf (3 async b128 per thread)
    auto issue = [&](int kk, int buf) {
        // kk is uniform and K1 % 32 == 0 -> scalar segment select
        const __bf16* asrc = (kk < K1) ? (a1p + kk) : (a2p + (kk - K1));
        async_ld16(ldsA[buf],  asrc);
        async_ld16(ldsB0[buf], w0p + kk);
        async_ld16(ldsB1[buf], w1p + kk);
    };

    const int nChunks = Ktot >> 5;
    issue(0, 0);

    for (int ci = 0; ci < nChunks; ++ci) {
        const int buf = ci & 1;
        wait_async0();          // my part of chunk ci is in LDS
        __syncthreads();        // everyone's chunk ci is in LDS; buf^1 is free
        if (ci + 1 < nChunks)
            issue((ci + 1) << 5, buf ^ 1);   // overlaps with compute below

        // ---- fragments ----
        // A (16-bit 16x32 layout): lanes 0-15 -> K 0-7 & 16-23,
        //                          lanes 16-31 -> K 8-15 & 24-31.
        v16bf af[2];
        #pragma unroll
        for (int i = 0; i < 2; ++i) {
            const u32x4* aq =
                (const u32x4*)(&Abuf[buf][(wave_m * 32 + i * 16 + ll) * 32]);
            Frag32B f;
            f.q[0] = aq[lh];
            f.q[1] = aq[lh + 2];
            af[i] = f.v;
        }
        // B: lanes 0-15 -> K 0-7 & 16-23 of column ll, lanes 16-31 -> K 8-15 & 24-31.
        v16bf bfc[2];
        #pragma unroll
        for (int j = 0; j < 2; ++j) {
            const u32x4* bq =
                (const u32x4*)(&Bbuf[buf][(wave_n * 32 + j * 16 + ll) * 32]);
            Frag32B f;
            f.q[0] = bq[lh];
            f.q[1] = bq[lh + 2];
            bfc[j] = f.v;
        }

        #pragma unroll
        for (int i = 0; i < 2; ++i)
            #pragma unroll
            for (int j = 0; j < 2; ++j)
                acc[i][j] = __builtin_amdgcn_wmma_f32_16x16x32_bf16(
                    false, af[i], false, bfc[j], (short)0, acc[i][j],
                    false, false);
    }

    // ---- epilogue: C/D layout VGPR r -> rows r (lanes 0-15) & r+8 (16-31) ----
    #pragma unroll
    for (int i = 0; i < 2; ++i) {
        #pragma unroll
        for (int j = 0; j < 2; ++j) {
            int m0  = mBase + wave_m * 32 + i * 16;
            int col = nBase + wave_n * 32 + j * 16 + ll;
            float bv = bias[col];
            #pragma unroll
            for (int r = 0; r < 8; ++r)
                Z[(size_t)(m0 + lh * 8 + r) * N4H + col] = acc[i][j][r] + bv;
        }
    }
}

// ---------------------------------------------------------------------------
// Gates + cell update + two LayerNorms.  One block per batch row.
// 256 threads x 2 elements of H=512.
// ---------------------------------------------------------------------------
__global__ void __launch_bounds__(256)
gates_ln_kernel(const float* __restrict__ z,     // [512][2048]
                float* __restrict__ c,           // [512][512] in/out
                __bf16* __restrict__ hb,         // [512][512] out (bf16)
                float*  __restrict__ hf,         // [512][512] out (f32)
                const float* __restrict__ gcg, const float* __restrict__ gcb,
                const float* __restrict__ ghg, const float* __restrict__ ghb)
{
    const int row = blockIdx.x;
    const int tid = threadIdx.x;
    __shared__ float s1[256];
    __shared__ float s2[256];

    const float* zr = z + (size_t)row * N4H;

    float cn[2], og[2];
    #pragma unroll
    for (int u = 0; u < 2; ++u) {
        int h = tid + u * 256;
        float zi = zr[h];
        float zf = zr[HH + h];
        float zo = zr[2 * HH + h];
        float zc = zr[3 * HH + h];
        float ig = __expf(zi);
        float fg = __expf(zf);
        og[u]    = 1.f / (1.f + __expf(-zo));
        float ct = tanhf(zc);
        cn[u] = fg * c[(size_t)row * HH + h] + ig * ct;
    }

    // ---- LN over cn ----
    s1[tid] = cn[0] + cn[1];
    s2[tid] = cn[0] * cn[0] + cn[1] * cn[1];
    __syncthreads();
    for (int off = 128; off > 0; off >>= 1) {
        if (tid < off) { s1[tid] += s1[tid + off]; s2[tid] += s2[tid + off]; }
        __syncthreads();
    }
    float mu = s1[0] * (1.f / HH);
    float rs = rsqrtf(s2[0] * (1.f / HH) - mu * mu + EPSL);
    __syncthreads();

    float hv[2];
    #pragma unroll
    for (int u = 0; u < 2; ++u) {
        int h = tid + u * 256;
        float cne = (cn[u] - mu) * rs * gcg[h] + gcb[h];
        c[(size_t)row * HH + h] = cne;
        hv[u] = og[u] * tanhf(cne);
    }

    // ---- LN over hv ----
    s1[tid] = hv[0] + hv[1];
    s2[tid] = hv[0] * hv[0] + hv[1] * hv[1];
    __syncthreads();
    for (int off = 128; off > 0; off >>= 1) {
        if (tid < off) { s1[tid] += s1[tid + off]; s2[tid] += s2[tid + off]; }
        __syncthreads();
    }
    float mu2 = s1[0] * (1.f / HH);
    float rs2 = rsqrtf(s2[0] * (1.f / HH) - mu2 * mu2 + EPSL);

    #pragma unroll
    for (int u = 0; u < 2; ++u) {
        int h = tid + u * 256;
        float he = (hv[u] - mu2) * rs2 * ghg[h] + ghb[h];
        hb[(size_t)row * HH + h] = (__bf16)he;
        hf[(size_t)row * HH + h] = he;
    }
}

// ---------------------------------------------------------------------------
// out = LN(h1) @ Wh + bh.   One block per batch row.
// ---------------------------------------------------------------------------
__global__ void __launch_bounds__(256)
final_kernel(const float* __restrict__ h1f,
             const float* __restrict__ lng, const float* __restrict__ lnb,
             const float* __restrict__ Wh,  const float* __restrict__ bh,
             float* __restrict__ out)
{
    const int row = blockIdx.x;
    const int tid = threadIdx.x;
    __shared__ float sv[HH];
    __shared__ float s1[256];
    __shared__ float s2[256];

    float v[2];
    #pragma unroll
    for (int u = 0; u < 2; ++u)
        v[u] = h1f[(size_t)row * HH + tid + u * 256];

    s1[tid] = v[0] + v[1];
    s2[tid] = v[0] * v[0] + v[1] * v[1];
    __syncthreads();
    for (int off = 128; off > 0; off >>= 1) {
        if (tid < off) { s1[tid] += s1[tid + off]; s2[tid] += s2[tid + off]; }
        __syncthreads();
    }
    float mu = s1[0] * (1.f / HH);
    float rs = rsqrtf(s2[0] * (1.f / HH) - mu * mu + EPSL);
    __syncthreads();

    #pragma unroll
    for (int u = 0; u < 2; ++u) {
        int h = tid + u * 256;
        sv[h] = (v[u] - mu) * rs * lng[h] + lnb[h];
    }
    __syncthreads();

    // 64 outputs, 4 K-segments of 128 each
    int d   = tid & 63;
    int seg = tid >> 6;
    float p = 0.f;
    #pragma unroll 4
    for (int k = seg * 128; k < seg * 128 + 128; ++k)
        p += sv[k] * Wh[(size_t)k * DD + d];
    s1[tid] = p;
    __syncthreads();
    if (tid < 64)
        out[(size_t)row * DD + tid] =
            s1[tid] + s1[tid + 64] + s1[tid + 128] + s1[tid + 192] + bh[tid];
}

// ---------------------------------------------------------------------------
// Host-side launch
// ---------------------------------------------------------------------------
extern "C" void kernel_launch(void* const* d_in, const int* in_sizes, int n_in,
                              void* d_out, int out_size, void* d_ws, size_t ws_size,
                              hipStream_t stream)
{
    const float* x    = (const float*)d_in[0];
    const float* W0   = (const float*)d_in[1];
    const float* b0   = (const float*)d_in[2];
    const float* gc0g = (const float*)d_in[3];
    const float* gc0b = (const float*)d_in[4];
    const float* gh0g = (const float*)d_in[5];
    const float* gh0b = (const float*)d_in[6];
    const float* W1   = (const float*)d_in[7];
    const float* b1   = (const float*)d_in[8];
    const float* gc1g = (const float*)d_in[9];
    const float* gc1b = (const float*)d_in[10];
    const float* gh1g = (const float*)d_in[11];
    const float* gh1b = (const float*)d_in[12];
    const float* lng  = (const float*)d_in[13];
    const float* lnb  = (const float*)d_in[14];
    const float* Wh   = (const float*)d_in[15];
    const float* bh   = (const float*)d_in[16];
    float* out = (float*)d_out;

    // --- workspace carve-out (assumed 256B-aligned base) ---
    char*  w   = (char*)d_ws;
    size_t off = 0;
    auto carve = [&](size_t bytes) -> void* {
        void* p = w + off;
        off += (bytes + 255) & ~(size_t)255;
        return p;
    };
    __bf16* W0t = (__bf16*)carve((size_t)N4H * (DD + HH) * 2);   // [2048][576]
    __bf16* W1t = (__bf16*)carve((size_t)N4H * (2 * HH) * 2);    // [2048][1024]
    __bf16* xt  = (__bf16*)carve((size_t)SS * BB * DD * 2);      // [S][B][D]
    __bf16* h0b = (__bf16*)carve((size_t)BB * HH * 2);
    __bf16* h1b = (__bf16*)carve((size_t)BB * HH * 2);
    float*  c0  = (float*)carve((size_t)BB * HH * 4);
    float*  c1  = (float*)carve((size_t)BB * HH * 4);
    float*  h0f = (float*)carve((size_t)BB * HH * 4);
    float*  h1f = (float*)carve((size_t)BB * HH * 4);
    float*  z   = (float*)carve((size_t)BB * N4H * 4);
    (void)ws_size; (void)n_in; (void)in_sizes; (void)out_size;

    // --- one-time prep (every call; deterministic) ---
    {
        int tot0 = N4H * (DD + HH);
        cvt_w_kernel<<<(tot0 + 255) / 256, 256, 0, stream>>>(W0, W0t, DD + HH, N4H, tot0);
        int tot1 = N4H * (2 * HH);
        cvt_w_kernel<<<(tot1 + 255) / 256, 256, 0, stream>>>(W1, W1t, 2 * HH, N4H, tot1);
        int totx = SS * BB * DD;
        cvt_x_kernel<<<(totx + 255) / 256, 256, 0, stream>>>(x, xt);
        // zero h0b, h1b, c0, c1 (contiguous carve region)
        int zdw = (int)(((size_t)BB * HH * 2 * 2 + (size_t)BB * HH * 4 * 2) / 4);
        zero_kernel<<<(zdw + 255) / 256, 256, 0, stream>>>((unsigned int*)h0b, zdw);
    }

    dim3 ggrid(N4H / 128, BB / 64);   // (16, 8)

    for (int t = 0; t < SS; ++t) {
        // layer 0:  z = [x_t | h0] @ W0 + b0
        gemm_bf16_kernel<<<ggrid, 256, 0, stream>>>(
            xt + (size_t)t * BB * DD, DD, DD,     // A1 = x_t [512][64]
            h0b, HH,                              // A2 = h0  [512][512]
            W0t, b0, z, DD + HH);
        gates_ln_kernel<<<BB, 256, 0, stream>>>(z, c0, h0b, h0f,
                                                gc0g, gc0b, gh0g, gh0b);
        // layer 1:  z = [h0_new | h1] @ W1 + b1
        gemm_bf16_kernel<<<ggrid, 256, 0, stream>>>(
            h0b, HH, HH,                          // A1 = h0_new
            h1b, HH,                              // A2 = h1
            W1t, b1, z, 2 * HH);
        gates_ln_kernel<<<BB, 256, 0, stream>>>(z, c1, h1b, h1f,
                                                gc1g, gc1b, gh1g, gh1b);
    }

    final_kernel<<<BB, 256, 0, stream>>>(h1f, lng, lnb, Wh, bh, out);
}
